// Attention_8057358648090
// MI455X (gfx1250) — compile-verified, matrix operand order checked by reading, not run on hardware
//
#include <hip/hip_runtime.h>

typedef _Float16 h16;
typedef __attribute__((ext_vector_type(16))) _Float16 v16h;
typedef __attribute__((ext_vector_type(8)))  _Float16 v8h;
typedef __attribute__((ext_vector_type(8)))  float    v8f;

#define B_SZ 4
#define SEQ  2048
#define CDIM 1024
#define NH   16
#define NKV  4
#define HD   64
#define ROWS (B_SZ * SEQ)   /* 8192 */

// ---------------------------------------------------------------------------
// D = A(16x32 f16) * B(32x16 f16) + C(16x16 f32)   -> v_wmma_f32_16x16x32_f16
// ---------------------------------------------------------------------------
__device__ __forceinline__ v8f wmma16(v16h a, v16h b, v8f c) {
  return __builtin_amdgcn_wmma_f32_16x16x32_f16(
      /*neg_a=*/false, a, /*neg_b=*/false, b,
      /*c_mod=*/(short)0, c, /*reuse_a=*/false, /*reuse_b=*/false);
}

// Assemble a 16-element fragment from two contiguous 8-element halves
// (K 0..7 and K 16..23 per the ISA 16-bit A/B fragment layout).
__device__ __forceinline__ v16h join16(v8h lo, v8h hi) {
  v16h r;
#pragma unroll
  for (int i = 0; i < 8; ++i) { r[i] = lo[i]; r[i + 8] = hi[i]; }
  return r;
}

// --- CDNA5 async memory->LDS DMA (no VGPR round-trip, tracked by ASYNCcnt) --
__device__ __forceinline__ unsigned lds_addr32(const void* p) {
  // generic pointers to LDS carry the LDS offset in addr[31:0]
  return (unsigned)(unsigned long long)(uintptr_t)p;
}
__device__ __forceinline__ void async_ld_b128(unsigned lds, const void* g) {
  asm volatile("global_load_async_to_lds_b128 %0, %1, off"
               :: "v"(lds), "v"((unsigned long long)(uintptr_t)g)
               : "memory");
}
__device__ __forceinline__ void wait_async_le4() {
  asm volatile("s_wait_asynccnt 0x4" ::: "memory");
}
__device__ __forceinline__ void wait_async_0() {
  asm volatile("s_wait_asynccnt 0x0" ::: "memory");
}

// ---------------------------------------------------------------------------
// f32 -> f16 elementwise convert
// ---------------------------------------------------------------------------
__global__ void cvt_f16(const float* __restrict__ in, h16* __restrict__ out, int n) {
  int i = blockIdx.x * blockDim.x + threadIdx.x;
  if (i < n) out[i] = (h16)in[i];
}

// w [K,N] f32 -> wT [N,K] f16  (so GEMM B-operand loads are contiguous per lane)
__global__ void transpose_f16(const float* __restrict__ w, h16* __restrict__ wT,
                              int K, int N) {
  int i = blockIdx.x * blockDim.x + threadIdx.x;
  if (i < K * N) {
    int k = i / N, n = i - k * N;
    wT[(size_t)n * K + k] = (h16)w[i];
  }
}

// ---------------------------------------------------------------------------
// C[M,N] f32 = A[M,K] f16 * BT[N,K]^T f16
// block = 256 threads = 8 waves; block tile 128(M) x 128(N); K step 32.
// wave grid 4(M) x 2(N); each wave: 32x64 = 2x4 WMMA fragments.
// Double-buffered LDS, filled with global_load_async_to_lds_b128 so the DMA
// of tile t+1 overlaps the WMMAs of tile t (ASYNCcnt pipeline).  The final
// iteration is peeled so the steady-state loop is branch-free.
// LDS rows padded to 40 halfs (80B) -> conflict-free ds_load_b128.
// ---------------------------------------------------------------------------
__global__ __launch_bounds__(256) void gemm_wmma(
    const h16* __restrict__ A, const h16* __restrict__ BT,
    float* __restrict__ C, int Nn, int K) {
  __shared__ h16 As[2][128 * 40];
  __shared__ h16 Bs[2][128 * 40];
  const int tid  = threadIdx.x;
  const int wid  = tid >> 5, lane = tid & 31;
  const int l16  = lane & 15, hl = lane >> 4;
  const int wm   = wid >> 1,  wn = wid & 1;
  const long m0  = (long)blockIdx.x * 128;
  const long n0  = (long)blockIdx.y * 128;

  const int srow = tid >> 2;          // staging row within a 64-row pass
  const int scg  = (tid & 3) * 8;     // staging col group (8 halfs = 16B)

  // issue the 4 async DMA statements that stage one 128x32 A + 128x32 B tile
  auto issue_tile = [&](int buf, int k0) {
#pragma unroll
    for (int p = 0; p < 2; ++p) {
      int row = p * 64 + srow;
      async_ld_b128(lds_addr32(&As[buf][row * 40 + scg]),
                    &A[(m0 + row) * K + k0 + scg]);
      async_ld_b128(lds_addr32(&Bs[buf][row * 40 + scg]),
                    &BT[(n0 + row) * K + k0 + scg]);
    }
  };

  v8f acc[2][4] = {};
  auto compute_tile = [&](int buf) {
    v16h af[2], bf[4];
#pragma unroll
    for (int mf = 0; mf < 2; ++mf) {
      const h16* pa = &As[buf][(wm * 32 + mf * 16 + l16) * 40 + hl * 8];
      af[mf] = join16(*(const v8h*)pa, *(const v8h*)(pa + 16));
    }
#pragma unroll
    for (int nf = 0; nf < 4; ++nf) {
      const h16* pb = &Bs[buf][(wn * 64 + nf * 16 + l16) * 40 + hl * 8];
      bf[nf] = join16(*(const v8h*)pb, *(const v8h*)(pb + 16));
    }
#pragma unroll
    for (int mf = 0; mf < 2; ++mf)
#pragma unroll
      for (int nf = 0; nf < 4; ++nf)
        acc[mf][nf] = wmma16(af[mf], bf[nf], acc[mf][nf]);
  };

  const int nt = K / 32;              // >= 2 for all our shapes
  issue_tile(0, 0);
  // steady state: branch-free — always DMA tile t+1 while computing tile t
  for (int t = 0; t < nt - 1; ++t) {
    const int buf = t & 1;
    issue_tile(buf ^ 1, (t + 1) * 32);
    wait_async_le4();                 // tile t (first 4 async ops) has landed
    __syncthreads();                  // publish tile across all 8 waves
    compute_tile(buf);
    __syncthreads();                  // all reads done before buf reuse
  }
  // peeled tail
  wait_async_0();
  __syncthreads();
  compute_tile((nt - 1) & 1);

  // C/D fragment layout: element r -> row r + 8*(lane>=16), col lane&15.
#pragma unroll
  for (int mf = 0; mf < 2; ++mf)
#pragma unroll
    for (int nf = 0; nf < 4; ++nf)
#pragma unroll
      for (int r = 0; r < 8; ++r) {
        long row = m0 + wm * 32 + mf * 16 + r + 8 * hl;
        long col = n0 + wn * 64 + nf * 16 + l16;
        C[row * Nn + col] = acc[mf][nf][r];
      }
}

// ---------------------------------------------------------------------------
// RMS-norm + RoPE + convert to f16.  One wave per (row, head); lane l holds
// elements l and l+32 of the 64-wide head row == rotate-half pairing.
// mode 0: Q (norm+rope, scale = 1.5/sqrt(64));  mode 1: K (norm+rope);
// mode 2: V (pure convert + transpose to [b][hkv][d][n]).
// ---------------------------------------------------------------------------
__global__ __launch_bounds__(256) void postproc(
    const float* __restrict__ raw, h16* __restrict__ out,
    int heads, int mode, float scale) {
  const int wid  = threadIdx.x >> 5, lane = threadIdx.x & 31;
  const long flat = (long)blockIdx.x * 8 + wid;  // row*heads + head
  const int row  = (int)(flat / heads);
  const int head = (int)(flat - (long)row * heads);
  const int b = row >> 11, n = row & (SEQ - 1);

  const float* p = raw + (size_t)row * heads * HD + head * HD;
  float x1 = p[lane], x2 = p[lane + 32];

  if (mode == 2) {  // V: transpose so P*V B-fragments load contiguously
    h16* vtp = out + ((size_t)(b * NKV + head) * HD) * SEQ + n;
    vtp[(size_t)lane * SEQ]        = (h16)x1;
    vtp[(size_t)(lane + 32) * SEQ] = (h16)x2;
    return;
  }

  float ss = x1 * x1 + x2 * x2;
#pragma unroll
  for (int m = 16; m >= 1; m >>= 1) ss += __shfl_xor(ss, m, 32);
  float rn = rsqrtf(ss * (1.0f / HD) + 1.1920929e-07f) * scale;

  // inv_freq = 10000^(-lane/32) = exp(-lane * ln(1e4)/32)
  float invf = __expf(-(float)lane * 0.28782313662f);
  float ang = (float)n * invf;
  float c = __cosf(ang), s = __sinf(ang);

  h16* q = out + (size_t)row * heads * HD + head * HD;
  q[lane]      = (h16)((x1 * c - x2 * s) * rn);
  q[lane + 32] = (h16)((x2 * c + x1 * s) * rn);
}

// ---------------------------------------------------------------------------
// Causal flash attention, one wave (=one block) per 16-row Q tile per head.
// KV chunks of 32: S = Q(16x64)*K^T via 4 WMMA, online softmax through LDS,
// O += P(16x32)*V via 4 WMMA.  Scale & gain pre-folded into Q.
// ---------------------------------------------------------------------------
__global__ __launch_bounds__(32) void flash_attn(
    const h16* __restrict__ qb, const h16* __restrict__ kb,
    const h16* __restrict__ vt, h16* __restrict__ ob) {
  __shared__ float S[16][33];
  __shared__ h16   P[16][40];
  __shared__ float alpha_s[16];

  const int lane = threadIdx.x;
  const int l16 = lane & 15, hl = lane >> 4;
  const int qt = blockIdx.x & 127;
  const int bh = blockIdx.x >> 7;
  const int h = bh & 15, b = bh >> 4;
  const int hkv = h >> 2;           // GQA: repeat_interleave groups of 4
  const int qr0 = qt * 16;

  // Q A-fragments (row = lane&15, K halves split by lane half)
  v16h qf[2];
  {
    const h16* qrow = qb + (size_t)(b * SEQ + qr0 + l16) * CDIM + h * HD;
#pragma unroll
    for (int t = 0; t < 2; ++t) {
      const h16* pq = qrow + t * 32 + hl * 8;
      qf[t] = join16(*(const v8h*)pq, *(const v8h*)(pq + 16));
    }
  }

  v8f o[4] = {};
  float m_r = -1e30f, s_r = 0.f;    // row state, duplicated in lane pairs
  const int rowg = qr0 + l16;
  const int kv_len = qr0 + 16;      // causal bound for this tile

  for (int kv0 = 0; kv0 < kv_len; kv0 += 32) {
    // hint next KV chunk toward the caches while we compute this one
    if (kv0 + 32 < kv_len) {
      int kvn = kv0 + 32 + l16;
      __builtin_prefetch(kb + (size_t)(b * SEQ + kvn) * (NKV * HD) + hkv * HD, 0, 3);
    }
    // ---- scores: two 16x16 tiles over this 32-wide kv chunk
#pragma unroll
    for (int j = 0; j < 2; ++j) {
      int kvc = kv0 + j * 16 + l16;
      if (kvc > SEQ - 1) kvc = SEQ - 1;   // defensive clamp
      const h16* krow = kb + (size_t)(b * SEQ + kvc) * (NKV * HD) + hkv * HD;
      v8f sacc = {};
#pragma unroll
      for (int t = 0; t < 2; ++t) {
        const h16* pk = krow + t * 32 + hl * 8;
        v16h kf = join16(*(const v8h*)pk, *(const v8h*)(pk + 16));
        sacc = wmma16(qf[t], kf, sacc);
      }
#pragma unroll
      for (int r = 0; r < 8; ++r)
        S[r + 8 * hl][j * 16 + l16] = sacc[r];
    }
    __syncthreads();

    // ---- online softmax: row = lane&15, each lane-pair half scans 16 cols
    float cmax = -1e30f;
#pragma unroll
    for (int c = 0; c < 16; ++c) {
      int col = hl * 16 + c, kvg = kv0 + col;
      float v = (kvg <= rowg) ? S[l16][col] : -1e30f;
      cmax = fmaxf(cmax, v);
    }
    cmax = fmaxf(cmax, __shfl_xor(cmax, 16, 32));
    float mnew = fmaxf(m_r, cmax);
    float csum = 0.f;
#pragma unroll
    for (int c = 0; c < 16; ++c) {
      int col = hl * 16 + c, kvg = kv0 + col;
      float pv = (kvg <= rowg) ? __expf(S[l16][col] - mnew) : 0.f;
      csum += pv;
      P[l16][col] = (h16)pv;
    }
    csum += __shfl_xor(csum, 16, 32);
    float a = __expf(m_r - mnew);
    s_r = s_r * a + csum;
    m_r = mnew;
    if (hl == 0) alpha_s[l16] = a;
    __syncthreads();

    // ---- rescale accumulator by per-row alpha
    float ar[8];
#pragma unroll
    for (int r = 0; r < 8; ++r) ar[r] = alpha_s[r + 8 * hl];
#pragma unroll
    for (int g = 0; g < 4; ++g)
#pragma unroll
      for (int r = 0; r < 8; ++r) o[g][r] *= ar[r];

    // ---- O += P * V ; P as A-fragment from LDS, V^T contiguous from global
    const h16* pp = &P[l16][hl * 8];
    v16h pf = join16(*(const v8h*)pp, *(const v8h*)(pp + 16));
#pragma unroll
    for (int g = 0; g < 4; ++g) {
      const h16* vrow =
          vt + ((size_t)(b * NKV + hkv) * HD + g * 16 + l16) * SEQ + kv0 + hl * 8;
      v16h vf = join16(*(const v8h*)vrow, *(const v8h*)(vrow + 16));
      o[g] = wmma16(pf, vf, o[g]);
    }
  }

  // ---- epilogue: divide by row sums, emit f16 at [b*N+n, h*64+d]
  if (hl == 0) alpha_s[l16] = 1.0f / s_r;
  __syncthreads();
  float inv[8];
#pragma unroll
  for (int r = 0; r < 8; ++r) inv[r] = alpha_s[r + 8 * hl];
#pragma unroll
  for (int g = 0; g < 4; ++g)
#pragma unroll
    for (int r = 0; r < 8; ++r)
      ob[(size_t)(b * SEQ + qr0 + r + 8 * hl) * CDIM + h * HD + g * 16 + l16] =
          (h16)(o[g][r] * inv[r]);
}

// ---------------------------------------------------------------------------
extern "C" void kernel_launch(void* const* d_in, const int* in_sizes, int n_in,
                              void* d_out, int out_size, void* d_ws, size_t ws_size,
                              hipStream_t stream) {
  const float* x  = (const float*)d_in[0];
  const float* wq = (const float*)d_in[1];
  const float* wk = (const float*)d_in[2];
  const float* wv = (const float*)d_in[3];
  const float* wo = (const float*)d_in[4];
  float* out = (float*)d_out;

  char* ws = (char*)d_ws;
  size_t off = 0;
  auto alloc = [&](size_t bytes) -> void* {
    void* p = ws + off;
    off = (off + bytes + 255) & ~(size_t)255;
    return p;
  };
  h16*   xb   = (h16*)alloc((size_t)ROWS * CDIM * 2);
  h16*   wqT  = (h16*)alloc((size_t)CDIM * CDIM * 2);
  h16*   wkT  = (h16*)alloc((size_t)CDIM * 256 * 2);
  h16*   wvT  = (h16*)alloc((size_t)CDIM * 256 * 2);
  h16*   woT  = (h16*)alloc((size_t)CDIM * CDIM * 2);
  float* qraw = (float*)alloc((size_t)ROWS * CDIM * 4);
  float* kraw = (float*)alloc((size_t)ROWS * 256 * 4);
  float* vraw = (float*)alloc((size_t)ROWS * 256 * 4);
  h16*   qb   = (h16*)alloc((size_t)ROWS * CDIM * 2);
  h16*   kb   = (h16*)alloc((size_t)ROWS * 256 * 2);
  h16*   vt   = (h16*)alloc((size_t)ROWS * 256 * 2 + 256);
  h16*   ob   = (h16*)qraw;  // qraw is dead after postproc -> reuse for O

  // 1) precision conversion + weight transposes (one-time, small)
  cvt_f16<<<(ROWS * CDIM + 255) / 256, 256, 0, stream>>>(x, xb, ROWS * CDIM);
  transpose_f16<<<(CDIM * CDIM + 255) / 256, 256, 0, stream>>>(wq, wqT, CDIM, CDIM);
  transpose_f16<<<(CDIM * 256  + 255) / 256, 256, 0, stream>>>(wk, wkT, CDIM, 256);
  transpose_f16<<<(CDIM * 256  + 255) / 256, 256, 0, stream>>>(wv, wvT, CDIM, 256);
  transpose_f16<<<(CDIM * CDIM + 255) / 256, 256, 0, stream>>>(wo, woT, CDIM, CDIM);

  // 2) QKV projections (async-DMA double-buffered WMMA GEMMs, f32 accumulate)
  gemm_wmma<<<dim3(ROWS / 128, CDIM / 128), 256, 0, stream>>>(xb, wqT, qraw, CDIM, CDIM);
  gemm_wmma<<<dim3(ROWS / 128, 256 / 128),  256, 0, stream>>>(xb, wkT, kraw, 256, CDIM);
  gemm_wmma<<<dim3(ROWS / 128, 256 / 128),  256, 0, stream>>>(xb, wvT, vraw, 256, CDIM);

  // 3) RMS-norm + RoPE (+ scale fold) ; V transpose
  postproc<<<ROWS * NH  / 8, 256, 0, stream>>>(qraw, qb, NH,  0, 0.1875f); // 1.5/sqrt(64)
  postproc<<<ROWS * NKV / 8, 256, 0, stream>>>(kraw, kb, NKV, 1, 1.0f);
  postproc<<<ROWS * NKV / 8, 256, 0, stream>>>(vraw, vt, NKV, 2, 1.0f);

  // 4) causal flash attention (WMMA), O in f16
  flash_attn<<<B_SZ * NH * (SEQ / 16), 32, 0, stream>>>(qb, kb, vt, ob);

  // 5) output projection -> f32 result
  gemm_wmma<<<dim3(ROWS / 128, CDIM / 128), 256, 0, stream>>>(ob, woT, out, CDIM, CDIM);
}
